// SSLLoss_8804682956809
// MI455X (gfx1250) — matrix operand order repaired and checked by reading, not compiled, not run on hardware
//
#include <hip/hip_runtime.h>
#include <hip/hip_bf16.h>
#include <math.h>

typedef __attribute__((ext_vector_type(16))) _Float16 v16h;
typedef __attribute__((ext_vector_type(8)))  _Float16 v8h;
typedef __attribute__((ext_vector_type(4)))  _Float16 v4h;
typedef __attribute__((ext_vector_type(8)))  float    v8f;
typedef __attribute__((ext_vector_type(4)))  float    v4f;

#define NROWS   8192
#define DDIM    128
#define NTILES  (NROWS / 16)          // 512
#define NCHUNK  16                    // column chunks per matrix
#define JT_PER_CHUNK (NTILES / NCHUNK) // 32 tiles (512 cols) per wave
// exp(x/0.1) = exp2(x * 10 * log2(e))
#define SCALE_E 14.4269504088896340736f

// Raw hardware exp2 (v_exp_f32). Safe here: all arguments are bounded to
// |x| <= ~14.5 (unit-norm dot products / 0.1 * log2e), far from the
// denormal/overflow ranges the OCML wrapper guards against.
__device__ __forceinline__ float fast_exp2(float x) {
#if defined(__has_builtin)
#if __has_builtin(__builtin_amdgcn_exp2f)
    return __builtin_amdgcn_exp2f(x);
#else
    return exp2f(x);
#endif
#else
    return exp2f(x);
#endif
}

// ---------------- workspace layout (bytes) ----------------
#define OFF_NF32  ((size_t)0)                                   // 4 * 8192*128 f32 = 16 MB
#define OFF_NF16  (OFF_NF32 + (size_t)4*NROWS*DDIM*4)           // 4 * 8192*128 f16 =  8 MB
#define OFF_P0    (OFF_NF16 + (size_t)4*NROWS*DDIM*2)           // 2*8192 f32
#define OFF_D11   (OFF_P0   + (size_t)2*NROWS*4)
#define OFF_D22   (OFF_D11  + (size_t)2*NROWS*4)
#define OFF_PART  (OFF_D22  + (size_t)2*NROWS*4)                // 6*16*8192 f32 = 3 MB
#define OFF_BLK   (OFF_PART + (size_t)6*NCHUNK*NROWS*4)         // 64 f32

// ---------------------------------------------------------------------------
// Stage A: row L2-normalize; emit fp32 normalized + f16 normalized copies.
// blockDim = (32, 8): one wave per row.
__global__ void __launch_bounds__(256)
ssl_normalize_kernel(const float* __restrict__ x0, const float* __restrict__ x1,
                     const float* __restrict__ x2, const float* __restrict__ x3,
                     float* __restrict__ nf32, _Float16* __restrict__ nf16) {
    const int lane = threadIdx.x;                       // 0..31
    const int row  = blockIdx.x * blockDim.y + threadIdx.y;
    const int mat  = blockIdx.y;                        // 0..3
    const float* src = (mat == 0) ? x0 : (mat == 1) ? x1 : (mat == 2) ? x2 : x3;

    const size_t base = (size_t)row * DDIM + lane * 4;
    v4f v = *(const v4f*)(src + base);
    float s = v[0]*v[0] + v[1]*v[1] + v[2]*v[2] + v[3]*v[3];
#pragma unroll
    for (int m = 1; m < 32; m <<= 1) s += __shfl_xor(s, m, 32);
    const float scale = 1.0f / fmaxf(sqrtf(s), 1e-12f);

    v4f o; v4h oh;
#pragma unroll
    for (int e = 0; e < 4; ++e) { o[e] = v[e] * scale; oh[e] = (_Float16)o[e]; }
    *(v4f*)(nf32 + (size_t)mat * NROWS * DDIM + base) = o;
    *(v4h*)(nf16 + (size_t)mat * NROWS * DDIM + base) = oh;
}

// ---------------------------------------------------------------------------
// Stage B: per pair/row: p0 = sum_d exp(n1*n2/T); diag terms exp(|n1|^2/T), exp(|n2|^2/T).
// blockDim = (32, 8): one wave per row; blockIdx.y = pair.
__global__ void __launch_bounds__(256)
ssl_pos_kernel(const float* __restrict__ nf32,
               float* __restrict__ p0, float* __restrict__ d11, float* __restrict__ d22) {
    const int lane = threadIdx.x;
    const int row  = blockIdx.x * blockDim.y + threadIdx.y;
    const int pair = blockIdx.y;                        // 0,1

    const size_t base = (size_t)row * DDIM + lane * 4;
    v4f a = *(const v4f*)(nf32 + (size_t)(2*pair)   * NROWS * DDIM + base);
    v4f b = *(const v4f*)(nf32 + (size_t)(2*pair+1) * NROWS * DDIM + base);

    float sp = 0.f, s1 = 0.f, s2 = 0.f;
#pragma unroll
    for (int e = 0; e < 4; ++e) {
        sp += fast_exp2(a[e] * b[e] * SCALE_E);
        s1 += a[e] * a[e];
        s2 += b[e] * b[e];
    }
#pragma unroll
    for (int m = 1; m < 32; m <<= 1) {
        sp += __shfl_xor(sp, m, 32);
        s1 += __shfl_xor(s1, m, 32);
        s2 += __shfl_xor(s2, m, 32);
    }
    if (lane == 0) {
        const int idx = pair * NROWS + row;
        p0[idx]  = sp;
        d11[idx] = fast_exp2(s1 * SCALE_E);
        d22[idx] = fast_exp2(s2 * SCALE_E);
    }
}

// ---------------------------------------------------------------------------
// Stage C: Gram tile GEMM (f16 WMMA, f32 acc) + exp + deterministic partial row sums.
// grid = (512 row-strips, 16 col-chunks, 6 matrices), block = 1 wave (32 threads).
// Matrix m: pair = m/3; kind 0: n1*n2^T, 1: n1*n1^T, 2: n2*n2^T (rows indexed by A).
__global__ void __launch_bounds__(32)
ssl_gram_rowsum_kernel(const _Float16* __restrict__ nf16, float* __restrict__ part) {
    const int lane  = threadIdx.x;
    const int itile = blockIdx.x;        // row strip (16 rows of A)
    const int chunk = blockIdx.y;        // column chunk (32 tiles = 512 cols of B)
    const int m     = blockIdx.z;        // 0..5
    const int pair  = m / 3, kind = m % 3;
    const int aMat  = (kind == 2) ? (2*pair + 1) : (2*pair);
    const int bMat  = (kind == 1) ? (2*pair)     : (2*pair + 1);
    const _Float16* A = nf16 + (size_t)aMat * NROWS * DDIM;
    const _Float16* B = nf16 + (size_t)bMat * NROWS * DDIM;

    const int mrow = lane & 15;          // M (for A) / N (for B) within tile
    const int half = lane >> 4;          // lane group 0/1

    // A operand (16x32 f16 per K-step), contiguous k order (B uses the same):
    // lane L: M = L%16; elems 0..15 = K[16*(L/16) .. +15]
    const _Float16* arow = A + (size_t)(itile * 16 + mrow) * DDIM;
    const int abase = half * 16;
    v16h aop[4];
#pragma unroll
    for (int kk = 0; kk < 4; ++kk) {
        v8h lo = *(const v8h*)(arow + kk * 32 + abase);
        v8h hi = *(const v8h*)(arow + kk * 32 + 8 + abase);
        aop[kk] = __builtin_shufflevector(lo, hi, 0,1,2,3,4,5,6,7,8,9,10,11,12,13,14,15);
    }

    // B operand (32x16 f16 per K-step): lane L: N = L%16; elems 0..15 = K[16*(L/16) .. +15]
    const int kbase = half * 16;

    float acc[8];
#pragma unroll
    for (int v = 0; v < 8; ++v) acc[v] = 0.f;

    for (int jt = 0; jt < JT_PER_CHUNK; ++jt) {
        const int j = chunk * JT_PER_CHUNK + jt;
        const _Float16* brow = B + (size_t)(j * 16 + mrow) * DDIM;
        v8f c = {};
#pragma unroll
        for (int kk = 0; kk < 4; ++kk) {
            v16h bop = *(const v16h*)(brow + kk * 32 + kbase);
            c = __builtin_amdgcn_wmma_f32_16x16x32_f16(
                    false, aop[kk], false, bop, (short)0, c, false, false);
        }
        // C layout: VGPR v, lanes 0-15 -> row v, lanes 16-31 -> row v+8; col = lane%16
#pragma unroll
        for (int v = 0; v < 8; ++v)
            acc[v] += fast_exp2(c[v] * SCALE_E);
    }

    // Sum over the 16 columns held by each 16-lane group (xor masks 1,2,4,8 stay in-group).
#pragma unroll
    for (int v = 0; v < 8; ++v) {
#pragma unroll
        for (int msk = 1; msk < 16; msk <<= 1)
            acc[v] += __shfl_xor(acc[v], msk, 32);
    }
    if (mrow == 0) {   // lane 0 -> rows 0..7, lane 16 -> rows 8..15
        float* out = part + ((size_t)m * NCHUNK + chunk) * NROWS + itile * 16 + half * 8;
#pragma unroll
        for (int v = 0; v < 8; ++v) out[v] = acc[v];
    }
}

// ---------------------------------------------------------------------------
// Stage D: per (pair,row) loss terms; 64 blocks x 256 threads, one item each.
__global__ void __launch_bounds__(256)
ssl_loss_partial_kernel(const float* __restrict__ p0a, const float* __restrict__ d11a,
                        const float* __restrict__ d22a, const float* __restrict__ part,
                        float* __restrict__ blk) {
    const int tid  = threadIdx.x;
    const int idx  = blockIdx.x * 256 + tid;   // 0..16383
    const int pair = idx >> 13;
    const int i    = idx & (NROWS - 1);
    const int m12 = pair * 3 + 0, m11 = pair * 3 + 1, m22 = pair * 3 + 2;

    float s12 = 0.f, s11 = 0.f, s22 = 0.f;
#pragma unroll
    for (int c = 0; c < NCHUNK; ++c) {
        s12 += part[((size_t)m12 * NCHUNK + c) * NROWS + i];
        s11 += part[((size_t)m11 * NCHUNK + c) * NROWS + i];
        s22 += part[((size_t)m22 * NCHUNK + c) * NROWS + i];
    }
    const int ri = pair * NROWS + i;
    const float p0 = p0a[ri];
    s11 -= d11a[ri];
    s22 -= d22a[ri];
    const float lp0 = logf(p0);
    const float l = (logf(s12 + s11 + p0) - lp0) + (logf(s12 + s22 + p0) - lp0);

    __shared__ float red[256];
    red[tid] = l;
    __syncthreads();
#pragma unroll
    for (int s = 128; s > 0; s >>= 1) {
        if (tid < s) red[tid] += red[tid + s];
        __syncthreads();
    }
    if (tid == 0) blk[blockIdx.x] = red[0];
}

// ---------------------------------------------------------------------------
// Stage E: final deterministic sum of 64 partials, scale by 1/4.
__global__ void ssl_loss_final_kernel(const float* __restrict__ blk, float* __restrict__ out) {
    if (threadIdx.x == 0) {
        float s = 0.f;
        for (int k = 0; k < 64; ++k) s += blk[k];
        out[0] = s * 0.25f;
    }
}

// ---------------------------------------------------------------------------
extern "C" void kernel_launch(void* const* d_in, const int* in_sizes, int n_in,
                              void* d_out, int out_size, void* d_ws, size_t ws_size,
                              hipStream_t stream) {
    const float* x0 = (const float*)d_in[0];   // uemb1
    const float* x1 = (const float*)d_in[1];   // uemb2
    const float* x2 = (const float*)d_in[2];   // iemb1
    const float* x3 = (const float*)d_in[3];   // iemb2

    char* ws = (char*)d_ws;
    float*     nf32 = (float*)    (ws + OFF_NF32);
    _Float16*  nf16 = (_Float16*) (ws + OFF_NF16);
    float*     p0   = (float*)    (ws + OFF_P0);
    float*     d11  = (float*)    (ws + OFF_D11);
    float*     d22  = (float*)    (ws + OFF_D22);
    float*     part = (float*)    (ws + OFF_PART);
    float*     blk  = (float*)    (ws + OFF_BLK);

    dim3 waveRows(32, 8);
    ssl_normalize_kernel<<<dim3(NROWS / 8, 4), waveRows, 0, stream>>>(x0, x1, x2, x3, nf32, nf16);
    ssl_pos_kernel<<<dim3(NROWS / 8, 2), waveRows, 0, stream>>>(nf32, p0, d11, d22);
    ssl_gram_rowsum_kernel<<<dim3(NTILES, NCHUNK, 6), dim3(32), 0, stream>>>(nf16, part);
    ssl_loss_partial_kernel<<<dim3(64), dim3(256), 0, stream>>>(p0, d11, d22, part, blk);
    ssl_loss_final_kernel<<<dim3(1), dim3(64), 0, stream>>>(blk, (float*)d_out);
}
